// GroupedQueryAttention_50861002719475
// MI455X (gfx1250) — compile-verified, matrix operand order checked by reading, not run on hardware
//
#include <hip/hip_runtime.h>
#include <hip/hip_bf16.h>

typedef __attribute__((ext_vector_type(16))) __bf16 v16bf;
typedef __attribute__((ext_vector_type(8)))  float  v8f;
typedef __attribute__((ext_vector_type(4)))  int    v4i_t;

#define D_MODEL      4096
#define NUM_HEADS    32
#define NUM_KV_HEADS 8
#define HEAD_DIM     128
#define SEQ          2048
#define KV_DIM       (NUM_KV_HEADS * HEAD_DIM)   // 1024
#define WINDOW       4096
#define SCALE_F      0.08838834764831845f        // 1/sqrt(128)
#define NEG_BIG      -1.0e9f

#define AS1 __attribute__((address_space(1)))
#define AS3 __attribute__((address_space(3)))

#if defined(__has_builtin)
#if __has_builtin(__builtin_amdgcn_global_load_async_to_lds_b128) && \
    __has_builtin(__builtin_amdgcn_s_wait_asynccnt)
#define USE_ASYNC_COPY 1
#endif
#endif

// 16-byte global->LDS copy. Async (ASYNCcnt-tracked, no VGPR data) when the
// toolchain exposes the gfx1250 builtin; otherwise b128 load + ds_store_b128.
// Builtin signature (from hipcc diagnostic): (int4 AS1*, int4 AS3*, imm, imm).
__device__ __forceinline__ void cp16(__bf16* dst_lds, const __bf16* src_g) {
#ifdef USE_ASYNC_COPY
  __builtin_amdgcn_global_load_async_to_lds_b128(
      (AS1 v4i_t*)src_g, (AS3 v4i_t*)dst_lds, 0, 0);
#else
  *reinterpret_cast<uint4*>(dst_lds) = *reinterpret_cast<const uint4*>(src_g);
#endif
}
__device__ __forceinline__ void async_fence() {
#ifdef USE_ASYNC_COPY
  __builtin_amdgcn_s_wait_asynccnt(0);
#endif
}

// ---------------------------------------------------------------------------
// WMMA helpers (CDNA5: D = A(16x32 bf16) * B(32x16 bf16) + C(16x16 f32))
// ---------------------------------------------------------------------------
__device__ __forceinline__ v8f wmma_bf16(v16bf a, v16bf b, v8f c) {
  return __builtin_amdgcn_wmma_f32_16x16x32_bf16(
      false, a, false, b, (short)0, c, false, false);
}

// Fragment load per ISA 7.12.2 (lane L: row L&15; K pairs {2v,2v+1}+8*(L>=16)
// for v<4, {16+2(v-4),+1}+8*(L>=16) for v>=4). Stride must be even.
__device__ __forceinline__ v16bf load_frag_lds(const __bf16* base, int stride,
                                               int row, int hi) {
  union { v16bf v; unsigned u[8]; } f;
  const __bf16* p = base + row * stride + hi * 8;
#pragma unroll
  for (int i = 0; i < 4; ++i) {
    f.u[i]     = *reinterpret_cast<const unsigned*>(p + 2 * i);
    f.u[i + 4] = *reinterpret_cast<const unsigned*>(p + 16 + 2 * i);
  }
  return f.v;
}

// ---------------------------------------------------------------------------
// GEMM: C[M,N] = A[M,K] * Bt[N,K]^T   (bf16 in, f32 out)
// Block 256 thr / 8 waves; tile 128(M) x 64(N) x 64(K); wave -> 32x32.
// All LDS rows are 16B-aligned (stride 72 elems) so tile fills are pure b128.
// ---------------------------------------------------------------------------
__global__ __launch_bounds__(256)
void gemm_bf16wmma(const __bf16* __restrict__ A, const __bf16* __restrict__ Bt,
                   float* __restrict__ C, int M, int N, int K) {
  __shared__ __align__(16) __bf16 sA[128 * 72];
  __shared__ __align__(16) __bf16 sB[64 * 72];

  const int tid  = threadIdx.x;
  const int bm   = blockIdx.y * 128;
  const int bn   = blockIdx.x * 64;
  const int w    = tid >> 5, lane = tid & 31;
  const int hi   = lane >> 4, l16 = lane & 15;
  const int wm   = w >> 1, wn = w & 1;

  v8f acc[2][2];
#pragma unroll
  for (int mt = 0; mt < 2; ++mt)
#pragma unroll
    for (int nt = 0; nt < 2; ++nt)
#pragma unroll
      for (int r = 0; r < 8; ++r) acc[mt][nt][r] = 0.0f;

  for (int k0 = 0; k0 < K; k0 += 64) {
    __syncthreads();
    // A tile 128x64 bf16: 1024 16B-chunks, 4 per thread
#pragma unroll
    for (int c = 0; c < 4; ++c) {
      const int id = c * 256 + tid;
      const int row = id >> 3, c8 = (id & 7) * 8;
      cp16(sA + row * 72 + c8, A + (size_t)(bm + row) * K + k0 + c8);
    }
    // B tile 64x64 bf16 (rows of Bt): 512 chunks, 2 per thread
#pragma unroll
    for (int c = 0; c < 2; ++c) {
      const int id = c * 256 + tid;
      const int row = id >> 3, c8 = (id & 7) * 8;
      cp16(sB + row * 72 + c8, Bt + (size_t)(bn + row) * K + k0 + c8);
    }
    if (k0 + 64 < K) {
      __builtin_prefetch(A + (size_t)(bm + (tid >> 1)) * K + k0 + 64, 0, 1);
      __builtin_prefetch(Bt + (size_t)(bn + (tid & 63)) * K + k0 + 64, 0, 1);
    }
    async_fence();
    __syncthreads();

#pragma unroll
    for (int ks = 0; ks < 2; ++ks) {
      v16bf afr[2], bfr[2];
#pragma unroll
      for (int mt = 0; mt < 2; ++mt)
        afr[mt] = load_frag_lds(sA + ks * 32, 72, wm * 32 + mt * 16 + l16, hi);
#pragma unroll
      for (int nt = 0; nt < 2; ++nt)
        bfr[nt] = load_frag_lds(sB + ks * 32, 72, wn * 32 + nt * 16 + l16, hi);
#pragma unroll
      for (int mt = 0; mt < 2; ++mt)
#pragma unroll
        for (int nt = 0; nt < 2; ++nt)
          acc[mt][nt] = wmma_bf16(afr[mt], bfr[nt], acc[mt][nt]);
    }
  }

#pragma unroll
  for (int mt = 0; mt < 2; ++mt)
#pragma unroll
    for (int nt = 0; nt < 2; ++nt) {
      const int row0 = bm + wm * 32 + mt * 16 + hi * 8;
      const int col  = bn + wn * 32 + nt * 16 + l16;
#pragma unroll
      for (int r = 0; r < 8; ++r)
        C[(size_t)(row0 + r) * N + col] = acc[mt][nt][r];
    }
}

// ---------------------------------------------------------------------------
// Conversion / layout kernels
// ---------------------------------------------------------------------------
// Flat f32 -> bf16 with scale (4 elements / thread)
__global__ void cvt_scale_kernel(const float* __restrict__ src,
                                 __bf16* __restrict__ dst, float scale, int n4) {
  const int i = blockIdx.x * blockDim.x + threadIdx.x;
  if (i >= n4) return;
  const float4 f = reinterpret_cast<const float4*>(src)[i];
  union { __bf16 h[4]; uint2 u; } o;
  o.h[0] = (__bf16)(f.x * scale);
  o.h[1] = (__bf16)(f.y * scale);
  o.h[2] = (__bf16)(f.z * scale);
  o.h[3] = (__bf16)(f.w * scale);
  reinterpret_cast<uint2*>(dst)[i] = o.u;
}

// W[K,N] f32 -> Wt[N,K] bf16 (32x32 LDS tile transpose, 256 threads)
__global__ __launch_bounds__(256)
void cvt_transpose_kernel(const float* __restrict__ W, __bf16* __restrict__ Wt,
                          int K, int N) {
  __shared__ float tile[32][33];
  const int k0 = blockIdx.y * 32, n0 = blockIdx.x * 32;
  const int tx = threadIdx.x & 31, ty = threadIdx.x >> 5; // 32 x 8
#pragma unroll
  for (int r = 0; r < 32; r += 8)
    tile[ty + r][tx] = W[(size_t)(k0 + ty + r) * N + n0 + tx];
  __syncthreads();
#pragma unroll
  for (int r = 0; r < 32; r += 8)
    Wt[(size_t)(n0 + ty + r) * K + k0 + tx] = (__bf16)tile[tx][ty + r];
}

// RoPE on q (in place, f32). q layout: [s][h*128 + d]
__global__ void rope_q_kernel(float* __restrict__ q, const int* __restrict__ offp) {
  const int idx = blockIdx.x * blockDim.x + threadIdx.x; // S*H*64
  if (idx >= SEQ * NUM_HEADS * 64) return;
  const int d = idx & 63;
  const int h = (idx >> 6) & (NUM_HEADS - 1);
  const int s = idx >> 11;
  float* base = q + (size_t)s * D_MODEL + h * HEAD_DIM;
  const float pos = (float)(s + *offp);
  const float inv = __powf(10000.0f, -(float)d * (1.0f / 64.0f));
  float sn, cs;
  __sincosf(pos * inv, &sn, &cs);
  const float x1 = base[d], x2 = base[d + 64];
  base[d]      = x1 * cs - x2 * sn;
  base[d + 64] = x2 * cs + x1 * sn;
}

// RoPE on k (in place) + scatter roped k (f32) to k_window [kvh][s][d]
__global__ void rope_k_kernel(float* __restrict__ k, float* __restrict__ k_win,
                              const int* __restrict__ offp) {
  const int idx = blockIdx.x * blockDim.x + threadIdx.x; // S*KVH*64
  if (idx >= SEQ * NUM_KV_HEADS * 64) return;
  const int d = idx & 63;
  const int h = (idx >> 6) & (NUM_KV_HEADS - 1);
  const int s = idx >> 9;
  float* base = k + (size_t)s * KV_DIM + h * HEAD_DIM;
  const float pos = (float)(s + *offp);
  const float inv = __powf(10000.0f, -(float)d * (1.0f / 64.0f));
  float sn, cs;
  __sincosf(pos * inv, &sn, &cs);
  const float x1 = base[d], x2 = base[d + 64];
  const float o1 = x1 * cs - x2 * sn;
  const float o2 = x2 * cs + x1 * sn;
  base[d] = o1;
  base[d + 64] = o2;
  float* wb = k_win + ((size_t)h * SEQ + s) * HEAD_DIM;
  wb[d] = o1;
  wb[d + 64] = o2;
}

// v_window [kvh][s][d] (f32) <- v [s][kvh*128 + d]
__global__ void copy_v_kernel(const float* __restrict__ v, float* __restrict__ v_win) {
  const int idx = blockIdx.x * blockDim.x + threadIdx.x; // S*KV_DIM
  if (idx >= SEQ * KV_DIM) return;
  const int d = idx & (HEAD_DIM - 1);
  const int h = (idx >> 7) & (NUM_KV_HEADS - 1);
  const int s = idx >> 10;
  v_win[((size_t)h * SEQ + s) * HEAD_DIM + d] =
      v[(size_t)s * KV_DIM + h * HEAD_DIM + d];
}

// ---------------------------------------------------------------------------
// Flash attention. Block = 128 thr (4 waves); one block per (q-tile, head).
// qb: bf16 pre-scaled [s][h*128+d]; kb: bf16 [s][kvh*128+d]; vT: bf16 [kvh*128+d][s].
// Output attnb: bf16 [s][h*128+d].
// ---------------------------------------------------------------------------
__global__ __launch_bounds__(128)
void attn_kernel(const __bf16* __restrict__ qb, const __bf16* __restrict__ kb,
                 const __bf16* __restrict__ vT, __bf16* __restrict__ attnb,
                 const int* __restrict__ offp) {
  __shared__ __align__(16) __bf16 sK[64 * 136];     // K tile [j][d] (also Q staging)
  __shared__ __align__(16) __bf16 sV[128 * 72];     // V tile [d][j]
  __shared__ __align__(16) __bf16 sP[4 * 16 * 68];  // per-wave P [m][j]

  const int tid  = threadIdx.x;
  const int w    = tid >> 5, lane = tid & 31;
  const int hi   = lane >> 4, l16 = lane & 15;
  const int h    = blockIdx.y;
  const int kvh  = h >> 2;                 // REPEATS = 4
  const int qbase = blockIdx.x * 64;
  const int off  = *offp;

  // Stage Q (already scaled, bf16) through sK; capture register fragments.
#pragma unroll
  for (int c = 0; c < 8; ++c) {
    const int id = c * 128 + tid;           // 1024 chunks
    const int row = id >> 4, c8 = (id & 15) * 8;
    cp16(sK + row * 136 + c8,
         qb + (size_t)(qbase + row) * D_MODEL + h * HEAD_DIM + c8);
  }
  async_fence();
  __syncthreads();
  v16bf qf[4];
#pragma unroll
  for (int kt = 0; kt < 4; ++kt)
    qf[kt] = load_frag_lds(sK + kt * 32, 136, w * 16 + l16, hi);
  __syncthreads();

  float mstate[8], lstate[8];
  v8f oacc[8];
#pragma unroll
  for (int r = 0; r < 8; ++r) { mstate[r] = -3.0e38f; lstate[r] = 0.0f; }
#pragma unroll
  for (int nt = 0; nt < 8; ++nt)
#pragma unroll
    for (int r = 0; r < 8; ++r) oacc[nt][r] = 0.0f;

  int lb = qbase + off - WINDOW + 1; if (lb < 0) lb = 0;
  int ub = qbase + 63 + off;          if (ub > SEQ - 1) ub = SEQ - 1;
  const int jstart = lb >> 6, jend = ub >> 6;

  for (int jb = jstart; jb <= jend; ++jb) {
    // K tile 64x128 and V tile 128x64, both contiguous 16B-chunk copies.
#pragma unroll
    for (int c = 0; c < 8; ++c) {
      const int id = c * 128 + tid;
      const int row = id >> 4, c8 = (id & 15) * 8;
      cp16(sK + row * 136 + c8,
           kb + (size_t)(jb * 64 + row) * KV_DIM + kvh * HEAD_DIM + c8);
    }
#pragma unroll
    for (int c = 0; c < 8; ++c) {
      const int id = c * 128 + tid;
      const int d = id >> 3, c8 = (id & 7) * 8;
      cp16(sV + d * 72 + c8,
           vT + (size_t)(kvh * HEAD_DIM + d) * SEQ + jb * 64 + c8);
    }
    async_fence();
    __syncthreads();

    // S = Q * K^T  (16 rows x 64 cols per wave)
    v8f sacc[4];
#pragma unroll
    for (int nt = 0; nt < 4; ++nt)
#pragma unroll
      for (int r = 0; r < 8; ++r) sacc[nt][r] = 0.0f;
#pragma unroll
    for (int kt = 0; kt < 4; ++kt)
#pragma unroll
      for (int nt = 0; nt < 4; ++nt) {
        v16bf kf = load_frag_lds(sK + kt * 32, 136, nt * 16 + l16, hi);
        sacc[nt] = wmma_bf16(qf[kt], kf, sacc[nt]);
      }

    // Causal + sliding-window mask
#pragma unroll
    for (int nt = 0; nt < 4; ++nt) {
      const int kvcol = jb * 64 + nt * 16 + l16;
#pragma unroll
      for (int r = 0; r < 8; ++r) {
        const int qpos = qbase + w * 16 + r + 8 * hi + off;
        const bool ok = (kvcol <= qpos) && (qpos - kvcol < WINDOW);
        if (!ok) sacc[nt][r] = NEG_BIG;
      }
    }

    // Online softmax (each row in one 16-lane half at one VGPR slot).
    float scalef[8];
#pragma unroll
    for (int r = 0; r < 8; ++r) {
      float mx = sacc[0][r];
#pragma unroll
      for (int nt = 1; nt < 4; ++nt) mx = fmaxf(mx, sacc[nt][r]);
#pragma unroll
      for (int sh = 1; sh < 16; sh <<= 1) mx = fmaxf(mx, __shfl_xor(mx, sh));
      const float mnew = fmaxf(mstate[r], mx);
      scalef[r] = __expf(mstate[r] - mnew);
      mstate[r] = mnew;
      float rs = 0.0f;
#pragma unroll
      for (int nt = 0; nt < 4; ++nt) {
        const float sv = sacc[nt][r];
        const float p  = (sv > -0.5e9f) ? __expf(sv - mnew) : 0.0f;
        sacc[nt][r] = p;
        rs += p;
      }
#pragma unroll
      for (int sh = 1; sh < 16; sh <<= 1) rs += __shfl_xor(rs, sh);
      lstate[r] = lstate[r] * scalef[r] + rs;
    }
#pragma unroll
    for (int nt = 0; nt < 8; ++nt)
#pragma unroll
      for (int r = 0; r < 8; ++r) oacc[nt][r] *= scalef[r];

    // P (bf16) -> per-wave LDS region
    __bf16* sPw = sP + w * 16 * 68;
#pragma unroll
    for (int nt = 0; nt < 4; ++nt)
#pragma unroll
      for (int r = 0; r < 8; ++r)
        sPw[(r + 8 * hi) * 68 + nt * 16 + l16] = (__bf16)sacc[nt][r];
    __syncthreads();

    // O += P * V   (2 K-steps of 32 kv positions)
#pragma unroll
    for (int kt = 0; kt < 2; ++kt) {
      v16bf pf = load_frag_lds(sPw + kt * 32, 68, l16, hi);
#pragma unroll
      for (int nt = 0; nt < 8; ++nt) {
        v16bf vf = load_frag_lds(sV + kt * 32, 72, nt * 16 + l16, hi);
        oacc[nt] = wmma_bf16(pf, vf, oacc[nt]);
      }
    }
    __syncthreads();
  }

  // Normalize and write attn output as bf16 [s][h*128 + d]
#pragma unroll
  for (int r = 0; r < 8; ++r) {
    const float invl = lstate[r] > 0.0f ? 1.0f / lstate[r] : 0.0f;
    const int srow = qbase + w * 16 + r + 8 * hi;
#pragma unroll
    for (int nt = 0; nt < 8; ++nt)
      attnb[(size_t)srow * D_MODEL + h * HEAD_DIM + nt * 16 + l16] =
          (__bf16)(oacc[nt][r] * invl);
  }
}

// ---------------------------------------------------------------------------
extern "C" void kernel_launch(void* const* d_in, const int* in_sizes, int n_in,
                              void* d_out, int out_size, void* d_ws, size_t ws_size,
                              hipStream_t stream) {
  const float* x   = (const float*)d_in[0];
  const float* Wq  = (const float*)d_in[1];
  const float* Wk  = (const float*)d_in[2];
  const float* Wv  = (const float*)d_in[3];
  const float* Wo  = (const float*)d_in[4];
  const int*   off = (const int*)d_in[5];

  // bf16 workspace
  __bf16* xb    = (__bf16*)d_ws;                            // 2048x4096
  __bf16* wqT   = xb   + (size_t)SEQ * D_MODEL;             // 4096x4096
  __bf16* wkT   = wqT  + (size_t)D_MODEL * D_MODEL;         // 1024x4096
  __bf16* wvT   = wkT  + (size_t)KV_DIM * D_MODEL;          // 1024x4096
  __bf16* woT   = wvT  + (size_t)KV_DIM * D_MODEL;          // 4096x4096
  __bf16* qbuf  = woT  + (size_t)D_MODEL * D_MODEL;         // 2048x4096 (scaled)
  __bf16* kbuf  = qbuf + (size_t)SEQ * D_MODEL;             // 2048x1024
  __bf16* vT    = kbuf + (size_t)SEQ * KV_DIM;              // 1024x2048
  __bf16* attnb = vT   + (size_t)KV_DIM * SEQ;              // 2048x4096
  // f32 workspace
  float* q_f32 = (float*)(attnb + (size_t)SEQ * D_MODEL);   // 2048x4096
  float* k_f32 = q_f32 + (size_t)SEQ * D_MODEL;             // 2048x1024
  float* v_f32 = k_f32 + (size_t)SEQ * KV_DIM;              // 2048x1024

  float* out   = (float*)d_out;                             // 2048x4096
  float* k_win = out + (size_t)SEQ * D_MODEL;               // 8x2048x128
  float* v_win = k_win + (size_t)SEQ * KV_DIM;              // 8x2048x128

  // One-time conversions (bandwidth-trivial vs GEMM streams)
  cvt_scale_kernel<<<(SEQ * D_MODEL / 4) / 256, 256, 0, stream>>>(
      x, xb, 1.0f, SEQ * D_MODEL / 4);
  cvt_transpose_kernel<<<dim3(D_MODEL / 32, D_MODEL / 32), 256, 0, stream>>>(
      Wq, wqT, D_MODEL, D_MODEL);
  cvt_transpose_kernel<<<dim3(KV_DIM / 32, D_MODEL / 32), 256, 0, stream>>>(
      Wk, wkT, D_MODEL, KV_DIM);
  cvt_transpose_kernel<<<dim3(KV_DIM / 32, D_MODEL / 32), 256, 0, stream>>>(
      Wv, wvT, D_MODEL, KV_DIM);
  cvt_transpose_kernel<<<dim3(D_MODEL / 32, D_MODEL / 32), 256, 0, stream>>>(
      Wo, woT, D_MODEL, D_MODEL);

  // QKV projections (bf16 WMMA)
  gemm_bf16wmma<<<dim3(D_MODEL / 64, SEQ / 128), 256, 0, stream>>>(
      xb, wqT, q_f32, SEQ, D_MODEL, D_MODEL);
  gemm_bf16wmma<<<dim3(KV_DIM / 64, SEQ / 128), 256, 0, stream>>>(
      xb, wkT, k_f32, SEQ, KV_DIM, D_MODEL);
  gemm_bf16wmma<<<dim3(KV_DIM / 64, SEQ / 128), 256, 0, stream>>>(
      xb, wvT, v_f32, SEQ, KV_DIM, D_MODEL);

  // RoPE (f32), kv-window outputs, bf16 repacks for attention
  rope_q_kernel<<<(SEQ * NUM_HEADS * 64) / 256, 256, 0, stream>>>(q_f32, off);
  cvt_scale_kernel<<<(SEQ * D_MODEL / 4) / 256, 256, 0, stream>>>(
      q_f32, qbuf, SCALE_F, SEQ * D_MODEL / 4);
  rope_k_kernel<<<(SEQ * NUM_KV_HEADS * 64) / 256, 256, 0, stream>>>(
      k_f32, k_win, off);
  cvt_scale_kernel<<<(SEQ * KV_DIM / 4) / 256, 256, 0, stream>>>(
      k_f32, kbuf, 1.0f, SEQ * KV_DIM / 4);
  copy_v_kernel<<<(SEQ * KV_DIM) / 256, 256, 0, stream>>>(v_f32, v_win);
  cvt_transpose_kernel<<<dim3(KV_DIM / 32, SEQ / 32), 256, 0, stream>>>(
      v_f32, vT, SEQ, KV_DIM);   // v[s][dkv] -> vT[dkv][s]

  // Flash attention (bf16 WMMA, online softmax)
  attn_kernel<<<dim3(SEQ / 64, NUM_HEADS), dim3(128), 0, stream>>>(
      qbuf, kbuf, vT, attnb, off);

  // Output projection
  gemm_bf16wmma<<<dim3(D_MODEL / 64, SEQ / 128), 256, 0, stream>>>(
      attnb, woT, out, SEQ, D_MODEL, D_MODEL);
}
